// GraphAttentionLayer_69930657514079
// MI455X (gfx1250) — matrix operand order repaired and checked by reading, not compiled, not run on hardware
//
#include <hip/hip_runtime.h>
#include <hip/hip_bf16.h>

// ---------------------------------------------------------------------------
// GraphAttentionLayer for MI455X (gfx1250, wave32, WMMA + async/TDM staging).
// Pipeline: bf16 convert -> QKV WMMA GEMMs -> scores WMMA (+bias/mask) ->
// row softmax (probs = output #2) -> ctx WMMA -> out-proj WMMA (+residual) ->
// LayerNorm (output #1).
// ---------------------------------------------------------------------------

typedef __attribute__((ext_vector_type(16))) __bf16 v16bf;
typedef __attribute__((ext_vector_type(8)))  __bf16 v8bf;
typedef __attribute__((ext_vector_type(8)))  float  v8f;

constexpr int kB = 2;
constexpr int kS = 2048;
constexpr int kE = 768;
constexpr int kH = 8;
constexpr int kD = 96;
constexpr int kM = kB * kS;          // 4096 rows (b,s)
constexpr float kLnEps = 1e-5f;

constexpr int BM = 64, BN = 64, BK = 32;
constexpr int CBN = 96;              // ctx kernel: full head dim as N
constexpr int LDP = 40;              // padded LDS row stride (elems, keeps 16B align)

// ---- gfx1250 async global->LDS + waits -------------------------------------
#define USE_ASYNC_LDS 1

__device__ __forceinline__ void async_copy_b128(const __bf16* g, __bf16* l) {
#if USE_ASYNC_LDS
  const unsigned lds = (unsigned)(size_t)l;                // LDS byte offset
  const unsigned long long ga = (unsigned long long)(size_t)g;
  asm volatile("global_load_async_to_lds_b128 %0, %1, off"
               :: "v"(lds), "v"(ga) : "memory");
#else
  *(v8bf*)l = *(const v8bf*)g;
#endif
}

__device__ __forceinline__ void wait_async() {
#if USE_ASYNC_LDS
  asm volatile("s_wait_asynccnt 0x0" ::: "memory");
#endif
}

__device__ __forceinline__ void wait_tensor() {
#if __has_builtin(__builtin_amdgcn_s_wait_tensorcnt)
  __builtin_amdgcn_s_wait_tensorcnt(0);
#else
  asm volatile("s_wait_tensorcnt 0x0" ::: "memory");
#endif
}

// ---- Tensor Data Mover: 2D bf16 tile (32 wide x 64 rows) with LDS padding --
#if __has_builtin(__builtin_amdgcn_tensor_load_to_lds)
#define HAVE_TDM 1
typedef __attribute__((ext_vector_type(4))) unsigned int u32x4;
typedef __attribute__((ext_vector_type(8))) int i32x8;
typedef __attribute__((ext_vector_type(4))) int i32x4;

// Loads tile_dim0=32 x tile_dim1=64 bf16 elements, row stride tensorStride
// (elements), into LDS rows of 64B data + 16B pad (matches LDP=40).
// D# layout per CDNA5 ISA 8.3/8.4. This toolchain's builtin takes 6 args
// (g0, g1, g2, g3, g4, cpol) -- groups 2..4 zeroed for a 2D tensor.
__device__ __forceinline__ void tdm_load_tile_bf16(const __bf16* gsrc,
                                                   unsigned ldsAddr,
                                                   unsigned tensorDim0,
                                                   unsigned tensorDim1,
                                                   unsigned tensorStride) {
  const unsigned long long ga = (unsigned long long)(size_t)gsrc;
  u32x4 g0;
  g0[0] = 1u;                                          // count=1 (valid, user)
  g0[1] = ldsAddr;                                     // lds_addr
  g0[2] = (unsigned)ga;                                // global_addr[31:0]
  g0[3] = (unsigned)((ga >> 32) & 0x1FFFFFFu) | (2u << 30);  // [56:32] | type=2
  i32x8 g1;
  g1[0] = (1 << 16)        // data_size = 2 bytes
        | (1 << 20)        // pad_enable
        | (3 << 22)        // pad_interval: 16 DWORDs (64B) between pads
        | (3 << 25);       // pad_amount: 4 DWORDs (16B)
  g1[1] = (int)((tensorDim0 & 0xFFFFu) << 16);                       // dim0 lo
  g1[2] = (int)(((tensorDim0 >> 16) & 0xFFFFu) |
                ((tensorDim1 & 0xFFFFu) << 16));                     // dim0 hi | dim1 lo
  g1[3] = (int)(((tensorDim1 >> 16) & 0xFFFFu) | (32u << 16));       // dim1 hi | tile_dim0
  g1[4] = (int)64u;                                                  // tile_dim1 (tile_dim2=0)
  g1[5] = (int)tensorStride;                                         // dim0_stride lo32
  g1[6] = 0;                                                         // dim0_stride hi16
  g1[7] = 0;
  i32x4 z4 = {0, 0, 0, 0};
  i32x8 z8 = {0, 0, 0, 0, 0, 0, 0, 0};
  __builtin_amdgcn_tensor_load_to_lds(g0, g1, z4, z4, z8, 0);
}
#else
#define HAVE_TDM 0
#endif

// ---- WMMA fragment helpers (wave32 layouts per CDNA5 ISA 7.12.2) -----------
__device__ __forceinline__ v16bf load_frag_a(const __bf16* As, int mBase, int lane) {
  const int m   = mBase + (lane & 15);
  const int kHi = (lane >> 4) << 3;                 // 0 or 8
  const __bf16* p = As + m * LDP + kHi;
  v8bf lo = *(const v8bf*)(p);
  v8bf hi = *(const v8bf*)(p + 16);
  v16bf a;
#pragma unroll
  for (int i = 0; i < 8; ++i) { a[i] = lo[i]; a[i + 8] = hi[i]; }
  return a;
}

__device__ __forceinline__ v16bf load_frag_b(const __bf16* Bt, int nBase, int lane) {
  const int n  = nBase + (lane & 15);
  const int k0 = (lane >> 4) << 3;                  // 0 or 8
  const __bf16* p = Bt + n * LDP + k0;
  v8bf lo = *(const v8bf*)(p);
  v8bf hi = *(const v8bf*)(p + 16);
  v16bf b;
#pragma unroll
  for (int i = 0; i < 8; ++i) { b[i] = lo[i]; b[i + 8] = hi[i]; }
  return b;
}

__device__ __forceinline__ v8f wmma_bf16(v16bf a, v16bf b, v8f c) {
  return __builtin_amdgcn_wmma_f32_16x16x32_bf16(false, a, false, b, (short)0, c,
                                                 false, false);
}

// ---- elementwise convert ---------------------------------------------------
__global__ void __launch_bounds__(256)
cvt_f32_to_bf16(const float* __restrict__ in, __bf16* __restrict__ out, int n) {
  int i = blockIdx.x * 256 + threadIdx.x;
  if (i < n) out[i] = (__bf16)in[i];
}

// ---- QKV projection: [4096,768] x [768,768] + bias -> bf16 [B,H,S,D] -------
__global__ void __launch_bounds__(256)
qkv_gemm(const __bf16* __restrict__ Abf, const __bf16* __restrict__ Wbf,
         const float* __restrict__ bias, __bf16* __restrict__ outHead) {
  __shared__ __align__(16) __bf16 As[2][BM * LDP];
  __shared__ __align__(16) __bf16 Bt[2][BN * LDP];
  const int tid = threadIdx.x, lane = tid & 31, wave = tid >> 5;
  const int mTile = blockIdx.x * BM, nTile = blockIdx.y * BN;
  const int nSub = wave & 3, mSub0 = (wave >> 2) * 2;
  v8f acc0 = {}, acc1 = {};

  auto stageA = [&](int buf, int k0) {
#if HAVE_TDM
    if (wave == 0)
      tdm_load_tile_bf16(Abf + (size_t)mTile * kE + k0,
                         (unsigned)(size_t)&As[buf][0], kE, kM, kE);
#else
    const int m = tid >> 2, kk = (tid & 3) * 8;
    async_copy_b128(Abf + (size_t)(mTile + m) * kE + k0 + kk,
                    &As[buf][m * LDP + kk]);
#endif
  };
  auto stageB = [&](int buf, int k0) {  // transpose W 32x64 -> Bt[n][k]
    const int kk = tid >> 3, n0 = (tid & 7) * 8;
    v8bf v = *(const v8bf*)(Wbf + (size_t)(k0 + kk) * kE + nTile + n0);
#pragma unroll
    for (int j = 0; j < 8; ++j) Bt[buf][(n0 + j) * LDP + kk] = v[j];
  };
  auto waitStage = [&]() {
#if HAVE_TDM
    if (wave == 0) wait_tensor();
#else
    wait_async();
#endif
  };

  stageA(0, 0);
  stageB(0, 0);
  waitStage();
  __syncthreads();

  constexpr int nIter = kE / BK;
  for (int it = 0; it < nIter; ++it) {
    const int cur = it & 1;
    if (it + 1 < nIter) {
      stageA(cur ^ 1, (it + 1) * BK);
      stageB(cur ^ 1, (it + 1) * BK);
    }
    v16bf b  = load_frag_b(&Bt[cur][0], nSub * 16, lane);
    v16bf a0 = load_frag_a(&As[cur][0], mSub0 * 16, lane);
    v16bf a1 = load_frag_a(&As[cur][0], (mSub0 + 1) * 16, lane);
    acc0 = wmma_bf16(a0, b, acc0);
    acc1 = wmma_bf16(a1, b, acc1);
    if (it + 1 < nIter) waitStage();
    __syncthreads();
  }

  const int nC = nTile + nSub * 16 + (lane & 15);
  const int h = nC / kD, d = nC % kD;
  const float bv = bias[nC];
#pragma unroll
  for (int i = 0; i < 8; ++i) {
    const int m0 = mTile + mSub0 * 16 + i + ((lane >> 4) << 3);
    {
      const int b_ = m0 >> 11, s = m0 & (kS - 1);
      outHead[(((size_t)b_ * kH + h) * kS + s) * kD + d] = (__bf16)(acc0[i] + bv);
    }
    const int m1 = m0 + 16;
    {
      const int b_ = m1 >> 11, s = m1 & (kS - 1);
      outHead[(((size_t)b_ * kH + h) * kS + s) * kD + d] = (__bf16)(acc1[i] + bv);
    }
  }
}

// ---- scores: Q[S,D] x K^T[D,S] / sqrt(D) + bias + mask -> raw logits -------
__global__ void __launch_bounds__(256)
scores_gemm(const __bf16* __restrict__ Qh, const __bf16* __restrict__ Kh,
            const float* __restrict__ distM, const float* __restrict__ angM,
            const float* __restrict__ amask,
            const float* __restrict__ dwp, const float* __restrict__ dbp,
            const float* __restrict__ awp, const float* __restrict__ abp,
            float* __restrict__ scores) {
  __shared__ __align__(16) __bf16 As[2][BM * LDP];
  __shared__ __align__(16) __bf16 Bt[2][BN * LDP];
  const int tid = threadIdx.x, lane = tid & 31, wave = tid >> 5;
  const int qTile = blockIdx.x * BM, kTile = blockIdx.y * BN;
  const int bh = blockIdx.z, b_ = bh >> 3;
  const __bf16* Qp = Qh + (size_t)bh * kS * kD;
  const __bf16* Kp = Kh + (size_t)bh * kS * kD;
  const int nSub = wave & 3, mSub0 = (wave >> 2) * 2;
  v8f acc0 = {}, acc1 = {};

  // Both tiles are straight copies -> async global->LDS.
  auto stage = [&](int buf, int k0) {
    const int m = tid >> 2, kk = (tid & 3) * 8;
    async_copy_b128(Qp + (size_t)(qTile + m) * kD + k0 + kk,
                    &As[buf][m * LDP + kk]);
    async_copy_b128(Kp + (size_t)(kTile + m) * kD + k0 + kk,
                    &Bt[buf][m * LDP + kk]);
  };

  stage(0, 0);
  wait_async();
  __syncthreads();

  constexpr int nIter = kD / BK;     // 3
  for (int it = 0; it < nIter; ++it) {
    const int cur = it & 1;
    if (it + 1 < nIter) stage(cur ^ 1, (it + 1) * BK);
    v16bf b  = load_frag_b(&Bt[cur][0], nSub * 16, lane);
    v16bf a0 = load_frag_a(&As[cur][0], mSub0 * 16, lane);
    v16bf a1 = load_frag_a(&As[cur][0], (mSub0 + 1) * 16, lane);
    acc0 = wmma_bf16(a0, b, acc0);
    acc1 = wmma_bf16(a1, b, acc1);
    if (it + 1 < nIter) wait_async();
    __syncthreads();
  }

  const float dw = dwp[0], aw = awp[0];
  const float cb = dbp[0] + abp[0];
  const float inv = 0.1020620726159658f;            // 1/sqrt(96)
  const int key = kTile + nSub * 16 + (lane & 15);
  const float mterm = (1.0f - amask[b_ * kS + key]) * -10000.0f + cb;
  float* outp = scores + (size_t)bh * kS * kS;
#pragma unroll
  for (int i = 0; i < 8; ++i) {
    const int q0 = qTile + mSub0 * 16 + i + ((lane >> 4) << 3);
    const size_t bi0 = ((size_t)b_ * kS + q0) * kS + key;
    outp[(size_t)q0 * kS + key] = acc0[i] * inv + dw * distM[bi0] + aw * angM[bi0] + mterm;
    const int q1 = q0 + 16;
    const size_t bi1 = ((size_t)b_ * kS + q1) * kS + key;
    outp[(size_t)q1 * kS + key] = acc1[i] * inv + dw * distM[bi1] + aw * angM[bi1] + mterm;
  }
}

// ---- row softmax over 2048 keys, in place ----------------------------------
__global__ void __launch_bounds__(256)
softmax_rows(float* __restrict__ p) {
  const int row = blockIdx.x, tid = threadIdx.x;
  float* x = p + (size_t)row * kS;
  float v[8];
  float m = -3.4e38f;
#pragma unroll
  for (int j = 0; j < 8; ++j) { v[j] = x[tid + j * 256]; m = fmaxf(m, v[j]); }
  __shared__ float sb[8];
#pragma unroll
  for (int o = 16; o > 0; o >>= 1) m = fmaxf(m, __shfl_xor(m, o, 32));
  if ((tid & 31) == 0) sb[tid >> 5] = m;
  __syncthreads();
  m = sb[0];
#pragma unroll
  for (int j = 1; j < 8; ++j) m = fmaxf(m, sb[j]);
  __syncthreads();
  float s = 0.f;
#pragma unroll
  for (int j = 0; j < 8; ++j) { v[j] = __expf(v[j] - m); s += v[j]; }
#pragma unroll
  for (int o = 16; o > 0; o >>= 1) s += __shfl_xor(s, o, 32);
  if ((tid & 31) == 0) sb[tid >> 5] = s;
  __syncthreads();
  s = 0.f;
#pragma unroll
  for (int j = 0; j < 8; ++j) s += sb[j];
  const float r = 1.0f / s;
#pragma unroll
  for (int j = 0; j < 8; ++j) x[tid + j * 256] = v[j] * r;
}

// ---- ctx: probs[S,S] x V[S,D] -> ctx f32 [B,S,E] ---------------------------
__global__ void __launch_bounds__(256)
ctx_gemm(const float* __restrict__ probs, const __bf16* __restrict__ Vh,
         float* __restrict__ ctx) {
  __shared__ __align__(16) __bf16 As[BM * LDP];     // 64 q x 32 k
  __shared__ __align__(16) __bf16 Bt[CBN * LDP];    // 96 n x 32 k
  const int tid = threadIdx.x, lane = tid & 31, wave = tid >> 5;
  const int qTile = blockIdx.x * BM;
  const int bh = blockIdx.y, b_ = bh >> 3, h = bh & 7;
  const float*  Pp = probs + (size_t)bh * kS * kS;
  const __bf16* Vp = Vh + (size_t)bh * kS * kD;
  const int mSub = wave >> 1;
  const int n3 = (wave & 1) * 3;
  v8f acc[3]; acc[0] = {}; acc[1] = {}; acc[2] = {};

  for (int k0 = 0; k0 < kS; k0 += BK) {
    { // probs tile 64x32 f32 -> bf16 in LDS
      const int m = tid >> 2, kk = (tid & 3) * 8;
      const float* g = Pp + (size_t)(qTile + m) * kS + k0 + kk;
      __bf16* dst = &As[m * LDP + kk];
#pragma unroll
      for (int j = 0; j < 8; ++j) dst[j] = (__bf16)g[j];
    }
    { // V tile 32x96 transposed to Bt[n][k]
      const int kk = tid >> 3, n0 = (tid & 7) * 12;
      const __bf16* g = Vp + (size_t)(k0 + kk) * kD + n0;
#pragma unroll
      for (int j = 0; j < 12; ++j) Bt[(n0 + j) * LDP + kk] = g[j];
    }
    if (k0 + BK < kS)
      __builtin_prefetch(Pp + (size_t)(qTile + (tid >> 2)) * kS + k0 + BK, 0, 1);
    __syncthreads();
    v16bf a = load_frag_a(As, mSub * 16, lane);
#pragma unroll
    for (int jj = 0; jj < 3; ++jj) {
      v16bf b = load_frag_b(Bt, (n3 + jj) * 16, lane);
      acc[jj] = wmma_bf16(a, b, acc[jj]);
    }
    __syncthreads();
  }
#pragma unroll
  for (int jj = 0; jj < 3; ++jj) {
    const int n = (n3 + jj) * 16 + (lane & 15);     // 0..95
#pragma unroll
    for (int i = 0; i < 8; ++i) {
      const int q = qTile + mSub * 16 + i + ((lane >> 4) << 3);
      ctx[((size_t)b_ * kS + q) * kE + h * kD + n] = acc[jj][i];
    }
  }
}

// ---- output projection + bias + residual -> res f32 ------------------------
__global__ void __launch_bounds__(256)
outproj_gemm(const __bf16* __restrict__ Abf, const __bf16* __restrict__ Wbf,
             const float* __restrict__ bias, const float* __restrict__ hidden,
             float* __restrict__ res) {
  __shared__ __align__(16) __bf16 As[2][BM * LDP];
  __shared__ __align__(16) __bf16 Bt[2][BN * LDP];
  const int tid = threadIdx.x, lane = tid & 31, wave = tid >> 5;
  const int mTile = blockIdx.x * BM, nTile = blockIdx.y * BN;
  const int nSub = wave & 3, mSub0 = (wave >> 2) * 2;
  v8f acc0 = {}, acc1 = {};

  auto stageA = [&](int buf, int k0) {
#if HAVE_TDM
    if (wave == 0)
      tdm_load_tile_bf16(Abf + (size_t)mTile * kE + k0,
                         (unsigned)(size_t)&As[buf][0], kE, kM, kE);
#else
    const int m = tid >> 2, kk = (tid & 3) * 8;
    async_copy_b128(Abf + (size_t)(mTile + m) * kE + k0 + kk,
                    &As[buf][m * LDP + kk]);
#endif
  };
  auto stageB = [&](int buf, int k0) {
    const int kk = tid >> 3, n0 = (tid & 7) * 8;
    v8bf v = *(const v8bf*)(Wbf + (size_t)(k0 + kk) * kE + nTile + n0);
#pragma unroll
    for (int j = 0; j < 8; ++j) Bt[buf][(n0 + j) * LDP + kk] = v[j];
  };
  auto waitStage = [&]() {
#if HAVE_TDM
    if (wave == 0) wait_tensor();
#else
    wait_async();
#endif
  };

  stageA(0, 0);
  stageB(0, 0);
  waitStage();
  __syncthreads();

  constexpr int nIter = kE / BK;
  for (int it = 0; it < nIter; ++it) {
    const int cur = it & 1;
    if (it + 1 < nIter) {
      stageA(cur ^ 1, (it + 1) * BK);
      stageB(cur ^ 1, (it + 1) * BK);
    }
    v16bf b  = load_frag_b(&Bt[cur][0], nSub * 16, lane);
    v16bf a0 = load_frag_a(&As[cur][0], mSub0 * 16, lane);
    v16bf a1 = load_frag_a(&As[cur][0], (mSub0 + 1) * 16, lane);
    acc0 = wmma_bf16(a0, b, acc0);
    acc1 = wmma_bf16(a1, b, acc1);
    if (it + 1 < nIter) waitStage();
    __syncthreads();
  }

  const int n = nTile + nSub * 16 + (lane & 15);
  const float bv = bias[n];
#pragma unroll
  for (int i = 0; i < 8; ++i) {
    const int m0 = mTile + mSub0 * 16 + i + ((lane >> 4) << 3);
    res[(size_t)m0 * kE + n] = acc0[i] + bv + hidden[(size_t)m0 * kE + n];
    const int m1 = m0 + 16;
    res[(size_t)m1 * kE + n] = acc1[i] + bv + hidden[(size_t)m1 * kE + n];
  }
}

// ---- LayerNorm over E=768 --------------------------------------------------
__global__ void __launch_bounds__(256)
layernorm_rows(const float* __restrict__ x, const float* __restrict__ gamma,
               const float* __restrict__ beta, float* __restrict__ out) {
  const int row = blockIdx.x, tid = threadIdx.x;
  const float* xr = x + (size_t)row * kE;
  float v[3];
  float s = 0.f;
#pragma unroll
  for (int j = 0; j < 3; ++j) { v[j] = xr[tid + j * 256]; s += v[j]; }
  __shared__ float sb[8];
#pragma unroll
  for (int o = 16; o > 0; o >>= 1) s += __shfl_xor(s, o, 32);
  if ((tid & 31) == 0) sb[tid >> 5] = s;
  __syncthreads();
  s = 0.f;
#pragma unroll
  for (int j = 0; j < 8; ++j) s += sb[j];
  const float mu = s * (1.0f / kE);
  __syncthreads();
  float q = 0.f;
#pragma unroll
  for (int j = 0; j < 3; ++j) { const float t = v[j] - mu; q += t * t; }
#pragma unroll
  for (int o = 16; o > 0; o >>= 1) q += __shfl_xor(q, o, 32);
  if ((tid & 31) == 0) sb[tid >> 5] = q;
  __syncthreads();
  q = 0.f;
#pragma unroll
  for (int j = 0; j < 8; ++j) q += sb[j];
  const float inv = rsqrtf(q * (1.0f / kE) + kLnEps);
  float* orow = out + (size_t)row * kE;
#pragma unroll
  for (int j = 0; j < 3; ++j) {
    const int e = tid + j * 256;
    orow[e] = (v[j] - mu) * inv * gamma[e] + beta[e];
  }
}

// ---------------------------------------------------------------------------
extern "C" void kernel_launch(void* const* d_in, const int* in_sizes, int n_in,
                              void* d_out, int out_size, void* d_ws, size_t ws_size,
                              hipStream_t stream) {
  const float* hidden = (const float*)d_in[0];
  const float* distM  = (const float*)d_in[1];
  const float* angM   = (const float*)d_in[2];
  const float* amask  = (const float*)d_in[3];
  const float* Wq = (const float*)d_in[4];
  const float* bq = (const float*)d_in[5];
  const float* Wk = (const float*)d_in[6];
  const float* bk = (const float*)d_in[7];
  const float* Wv = (const float*)d_in[8];
  const float* bv = (const float*)d_in[9];
  const float* dw = (const float*)d_in[10];
  const float* db = (const float*)d_in[11];
  const float* aw = (const float*)d_in[12];
  const float* ab = (const float*)d_in[13];
  const float* Wo = (const float*)d_in[14];
  const float* bo = (const float*)d_in[15];
  const float* gamma = (const float*)d_in[16];
  const float* beta  = (const float*)d_in[17];

  float* out   = (float*)d_out;                         // [B,S,E]
  float* probs = out + (size_t)kM * kE;                 // [B,H,S,S]

  char* ws = (char*)d_ws;
  size_t off = 0;
  auto alloc = [&](size_t bytes) -> void* {
    void* p = ws + off;
    off += (bytes + 255) & ~(size_t)255;
    return p;
  };
  __bf16* hs_bf  = (__bf16*)alloc((size_t)kM * kE * 2);
  __bf16* wq_bf  = (__bf16*)alloc((size_t)kE * kE * 2);
  __bf16* wk_bf  = (__bf16*)alloc((size_t)kE * kE * 2);
  __bf16* wv_bf  = (__bf16*)alloc((size_t)kE * kE * 2);
  __bf16* wo_bf  = (__bf16*)alloc((size_t)kE * kE * 2);
  __bf16* q_bf   = (__bf16*)alloc((size_t)kM * kE * 2); // [B,H,S,D]
  __bf16* k_bf   = (__bf16*)alloc((size_t)kM * kE * 2);
  __bf16* v_bf   = (__bf16*)alloc((size_t)kM * kE * 2);
  float*  ctx_f  = (float*)alloc((size_t)kM * kE * 4);  // [B,S,E]
  __bf16* ctx_bf = (__bf16*)alloc((size_t)kM * kE * 2);
  float*  res_f  = (float*)alloc((size_t)kM * kE * 4);

  const int nHS = kM * kE;          // 3,145,728
  const int nW  = kE * kE;          // 589,824

  cvt_f32_to_bf16<<<(nHS + 255) / 256, 256, 0, stream>>>(hidden, hs_bf, nHS);
  cvt_f32_to_bf16<<<(nW + 255) / 256, 256, 0, stream>>>(Wq, wq_bf, nW);
  cvt_f32_to_bf16<<<(nW + 255) / 256, 256, 0, stream>>>(Wk, wk_bf, nW);
  cvt_f32_to_bf16<<<(nW + 255) / 256, 256, 0, stream>>>(Wv, wv_bf, nW);
  cvt_f32_to_bf16<<<(nW + 255) / 256, 256, 0, stream>>>(Wo, wo_bf, nW);

  dim3 gProj(kM / BM, kE / BN);                         // 64 x 12
  qkv_gemm<<<gProj, 256, 0, stream>>>(hs_bf, wq_bf, bq, q_bf);
  qkv_gemm<<<gProj, 256, 0, stream>>>(hs_bf, wk_bf, bk, k_bf);
  qkv_gemm<<<gProj, 256, 0, stream>>>(hs_bf, wv_bf, bv, v_bf);

  dim3 gScore(kS / BM, kS / BN, kB * kH);               // 32 x 32 x 16
  scores_gemm<<<gScore, 256, 0, stream>>>(q_bf, k_bf, distM, angM, amask,
                                          dw, db, aw, ab, probs);

  softmax_rows<<<kB * kH * kS, 256, 0, stream>>>(probs);

  dim3 gCtx(kS / BM, kB * kH);                          // 32 x 16
  ctx_gemm<<<gCtx, 256, 0, stream>>>(probs, v_bf, ctx_f);

  cvt_f32_to_bf16<<<(nHS + 255) / 256, 256, 0, stream>>>(ctx_f, ctx_bf, nHS);

  outproj_gemm<<<gProj, 256, 0, stream>>>(ctx_bf, wo_bf, bo, hidden, res_f);

  layernorm_rows<<<kM, 256, 0, stream>>>(res_f, gamma, beta, out);
}
